// Seq2Seq_31181462569444
// MI455X (gfx1250) — matrix operand order repaired and checked by reading, not compiled, not run on hardware
//
#include <hip/hip_runtime.h>
#include <hip/hip_bf16.h>
#include <math.h>

typedef __attribute__((ext_vector_type(16))) _Float16 v16h;
typedef __attribute__((ext_vector_type(8)))  _Float16 v8h;
typedef __attribute__((ext_vector_type(8)))  float    v8f;

#define DEV __device__ __forceinline__

static constexpr int Bn = 64, Sn = 128, Wn = 16, Dn = 256, Hn = 512;
static constexpr int Gn = 1024, Cn = 1024;
static constexpr int G4 = 4 * Hn;   // 2048 LSTM gate width
static constexpr int G3 = 3 * Gn;   // 3072 GRU gate width
static constexpr int DH = Dn + Hn;  // 768 GRU input width

// ---------------------------------------------------------------------------
// WMMA helpers (CDNA5 16x16x32 f16 -> f32)
// ---------------------------------------------------------------------------
DEV v8f wmma_f16(v16h a, v16h b, v8f c) {
  // 8 args: (neg_a, A, neg_b, B, c_mod, C, reuse_a, reuse_b)
  return __builtin_amdgcn_wmma_f32_16x16x32_f16(false, a, false, b, (short)0, c,
                                                false, false);
}

// A fragment, 16x32 f16, row-major source with leading dim ld (halves).
// Lane l: m = l&15, khalf = l>>4; elems 0..7 = K kbase+8*kh.., elems 8..15 = +16.
DEV v16h load_a(const _Float16* base, int ld, int mbase, int kbase, int lane) {
  int m = mbase + (lane & 15);
  int kh = lane >> 4;
  const _Float16* p = base + (size_t)m * ld + kbase + kh * 8;
  v8h lo = *(const v8h*)p;
  v8h hi = *(const v8h*)(p + 16);
  v16h a;
#pragma unroll
  for (int i = 0; i < 8; ++i) { a[i] = lo[i]; a[i + 8] = hi[i]; }
  return a;
}

// B fragment from prepacked buffer: tile (kt,nt), lane l holds 16 contiguous halves.
DEV v16h load_b(const _Float16* Bp, int NT, int kt, int nt, int lane) {
  const _Float16* p = Bp + ((((size_t)kt * NT + nt) * 32 + lane) << 4);
  return *(const v16h*)p;
}

DEV float sigmoid_f(float x) { return 1.0f / (1.0f + __expf(-x)); }
DEV float tanh_f(float x) {
  float e = __expf(2.0f * x);
  return (e - 1.0f) / (e + 1.0f);
}

// ---------------------------------------------------------------------------
// Weight packing: W row-major [N,K] f32 -> B-fragment tiles f16.
// packed[((kt*NT+nt)*32 + lane)*16 + v] = W[nt*16 + (lane&15)][kt*32 + (lane>>4)*8 + (v&7) + (v>>3)*16]
// ---------------------------------------------------------------------------
__global__ void pack_b_kernel(const float* __restrict__ W, _Float16* __restrict__ out,
                              int N, int K) {
  int i = blockIdx.x * blockDim.x + threadIdx.x;
  if (i >= N * K) return;
  int v = i & 15;
  int l = (i >> 4) & 31;
  int tile = i >> 9;
  int NT = N >> 4;
  int nt = tile % NT;
  int kt = tile / NT;
  int n = nt * 16 + (l & 15);
  int k = kt * 32 + (l >> 4) * 8 + (v & 7) + (v >> 3) * 16;
  out[i] = (_Float16)W[(size_t)n * K + k];
}

__global__ void zero16_kernel(_Float16* p, int n) {
  int i = blockIdx.x * blockDim.x + threadIdx.x;
  if (i < n) p[i] = (_Float16)0.f;
}
__global__ void zero32_kernel(float* p, int n) {
  int i = blockIdx.x * blockDim.x + threadIdx.x;
  if (i < n) p[i] = 0.f;
}

// ---------------------------------------------------------------------------
// Word-embedding gather (t-major): xeF[(t*B+b)*D+d] = Ew[words[b,t]]
// xeB uses length-aware reversed index clip(words_num[b]-1-t, 0).
// ---------------------------------------------------------------------------
__global__ void gather_xe_kernel(const int* __restrict__ words,
                                 const int* __restrict__ words_num,
                                 const float* __restrict__ Ew,
                                 _Float16* __restrict__ xeF,
                                 _Float16* __restrict__ xeB) {
  int i = blockIdx.x * blockDim.x + threadIdx.x;
  if (i >= Sn * Bn * Dn) return;
  int d = i & (Dn - 1);
  int b = (i >> 8) & (Bn - 1);
  int t = i >> 14;
  xeF[i] = (_Float16)Ew[(size_t)words[b * Sn + t] * Dn + d];
  int tb = words_num[b] - 1 - t;
  if (tb < 0) tb = 0;
  xeB[i] = (_Float16)Ew[(size_t)words[b * Sn + tb] * Dn + d];
}

// ---------------------------------------------------------------------------
// WMMA GEMM: C[M,N] = A_f16[M,K] @ packedB[K,N] (+bias[N]).
// Block = (mtile, group of 32 n-tiles). The 16xK A strip is staged in LDS once
// (padded stride -> conflict-free ds_load_b128 fragments); each of the 8 waves
// computes 4 adjacent 16x16 tiles, reusing one A fragment per 4 WMMAs.
// Optional (t*B+b) -> (b*S+t) row remap for logits (d_out is [B,S,C]).
// ---------------------------------------------------------------------------
__global__ __launch_bounds__(256) void gemm_tile_kernel(
    const _Float16* __restrict__ A, const _Float16* __restrict__ Bp,
    const float* __restrict__ bias, _Float16* __restrict__ Of16,
    float* __restrict__ Of32, int M, int N, int K, int swapBS) {
  extern __shared__ _Float16 as[];  // [16][K+8]
  const int tid = threadIdx.x, lane = tid & 31, wv = tid >> 5;
  const int NT = N >> 4;
  const int NG = NT >> 5;  // n-tile groups of 32 per block
  const int mt = blockIdx.x / NG;
  const int ntg = blockIdx.x % NG;
  const int lda = K + 8;

  // stage A strip [16, K] into LDS (vectorized, coalesced)
  for (int v = tid; v < (16 * K) >> 3; v += 256) {
    int idx = v << 3;
    int r = idx / K, c = idx % K;
    *(v8h*)&as[r * lda + c] = *(const v8h*)&A[(size_t)(mt * 16 + r) * K + c];
  }
  __syncthreads();

  const int nt0 = ntg * 32 + wv * 4;
  v8f acc[4] = {v8f{}, v8f{}, v8f{}, v8f{}};
  const int KT = K >> 5;
  for (int kt = 0; kt < KT; ++kt) {
    v16h a = load_a(as, lda, 0, kt * 32, lane);
    acc[0] = wmma_f16(a, load_b(Bp, NT, kt, nt0 + 0, lane), acc[0]);
    acc[1] = wmma_f16(a, load_b(Bp, NT, kt, nt0 + 1, lane), acc[1]);
    acc[2] = wmma_f16(a, load_b(Bp, NT, kt, nt0 + 2, lane), acc[2]);
    acc[3] = wmma_f16(a, load_b(Bp, NT, kt, nt0 + 3, lane), acc[3]);
  }
#pragma unroll
  for (int j = 0; j < 4; ++j) {
    int n = (nt0 + j) * 16 + (lane & 15);
    float bv = bias ? bias[n] : 0.f;
#pragma unroll
    for (int r = 0; r < 8; ++r) {
      int m = mt * 16 + r + 8 * (lane >> 4);
      float v = acc[j][r] + bv;
      if (Of32) {
        int orow = swapBS ? ((m & (Bn - 1)) * Sn + (m >> 6)) : m;
        Of32[(size_t)orow * N + n] = v;
      } else {
        Of16[(size_t)m * N + n] = (_Float16)v;
      }
    }
  }
}

// ---------------------------------------------------------------------------
// Char LSTM, persistent: each WG owns 16 words for all 16 char steps.
// h in LDS (f16, ping-pong), c in VGPRs (f32). 8 waves x 4 hidden tiles cover
// H=512; each hidden tile computes 4 gate tiles (i,f,g,o) via WMMA.
// ---------------------------------------------------------------------------
__global__ __launch_bounds__(256) void char_lstm_kernel(
    const int* __restrict__ chars, const int* __restrict__ char_lens,
    const int* __restrict__ words_num, const float* __restrict__ Ec,
    const _Float16* __restrict__ WihP, const _Float16* __restrict__ WhhP,
    const float* __restrict__ bias, _Float16* __restrict__ cenc) {
  __shared__ _Float16 xs[16][Dn + 8];       // +8 halves: bank-conflict-free frags
  __shared__ _Float16 hs[2][16][Hn + 8];
  __shared__ int lens_s[16];

  const int tid = threadIdx.x, lane = tid & 31, wv = tid >> 5;
  const int r0 = blockIdx.x * 16;

  for (int i = tid; i < 16 * (Hn + 8); i += 256)
    hs[0][i / (Hn + 8)][i % (Hn + 8)] = (_Float16)0.f;
  if (tid < 16) lens_s[tid] = char_lens[r0 + tid];

  float creg[4][8];
#pragma unroll
  for (int j = 0; j < 4; ++j)
#pragma unroll
    for (int r = 0; r < 8; ++r) creg[j][r] = 0.f;

  int cur = 0;
  for (int t = 0; t < Wn; ++t) {
    __syncthreads();  // previous step's reads complete
    {                 // gather x_t = Ec[chars[:, t]] into LDS as f16
      int row = tid >> 4;
      int c0 = (tid & 15) << 4;
      int ci = chars[(size_t)(r0 + row) * Wn + t];
      const float* src = Ec + (size_t)ci * Dn + c0;
#pragma unroll
      for (int i = 0; i < 16; ++i) xs[row][c0 + i] = (_Float16)src[i];
    }
    __syncthreads();  // xs ready, hs[cur] stable

#pragma unroll
    for (int j = 0; j < 4; ++j) {
      int ht = wv + 8 * j;  // hidden tile 0..31
      v8f a0 = {}, a1 = {}, a2 = {}, a3 = {};
      for (int kt = 0; kt < Dn / 32; ++kt) {
        v16h a = load_a(&xs[0][0], Dn + 8, 0, kt * 32, lane);
        a0 = wmma_f16(a, load_b(WihP, G4 / 16, kt, 0 * 32 + ht, lane), a0);
        a1 = wmma_f16(a, load_b(WihP, G4 / 16, kt, 1 * 32 + ht, lane), a1);
        a2 = wmma_f16(a, load_b(WihP, G4 / 16, kt, 2 * 32 + ht, lane), a2);
        a3 = wmma_f16(a, load_b(WihP, G4 / 16, kt, 3 * 32 + ht, lane), a3);
      }
      for (int kt = 0; kt < Hn / 32; ++kt) {
        v16h a = load_a(&hs[cur][0][0], Hn + 8, 0, kt * 32, lane);
        a0 = wmma_f16(a, load_b(WhhP, G4 / 16, kt, 0 * 32 + ht, lane), a0);
        a1 = wmma_f16(a, load_b(WhhP, G4 / 16, kt, 1 * 32 + ht, lane), a1);
        a2 = wmma_f16(a, load_b(WhhP, G4 / 16, kt, 2 * 32 + ht, lane), a2);
        a3 = wmma_f16(a, load_b(WhhP, G4 / 16, kt, 3 * 32 + ht, lane), a3);
      }
      int nloc = ht * 16 + (lane & 15);
      float bi = bias[nloc], bf = bias[Hn + nloc];
      float bg = bias[2 * Hn + nloc], bo = bias[3 * Hn + nloc];
#pragma unroll
      for (int r = 0; r < 8; ++r) {
        int m = r + 8 * (lane >> 4);  // row within tile (C/D VGPR layout)
        bool live = t < lens_s[m];
        float iv = sigmoid_f(a0[r] + bi);
        float fv = sigmoid_f(a1[r] + bf);
        float gv = tanh_f(a2[r] + bg);
        float ov = sigmoid_f(a3[r] + bo);
        float c2 = fv * creg[j][r] + iv * gv;
        float h2 = ov * tanh_f(c2);
        float hold = (float)hs[cur][m][nloc];
        if (live) creg[j][r] = c2;
        hs[cur ^ 1][m][nloc] = (_Float16)(live ? h2 : hold);
      }
    }
    cur ^= 1;
  }
  __syncthreads();
  // write masked final hidden: cenc[row, :] = (s < words_num[b]) ? h : 0
  for (int i = tid; i < 16 * Hn; i += 256) {
    int m = i >> 9, n = i & (Hn - 1);
    int row = r0 + m;
    int bb = row >> 7;   // / S
    int ss = row & 127;  // % S
    _Float16 v = (ss < words_num[bb]) ? hs[cur][m][n] : (_Float16)0.f;
    cenc[(size_t)row * Hn + n] = v;
  }
}

// ---------------------------------------------------------------------------
// Word bi-LSTM recurrence step (x-part precomputed in XF/XB).
// 256 waves: (dir, mtile, htile). h ping-pong in global (f16), c in f32.
// ---------------------------------------------------------------------------
__global__ __launch_bounds__(256) void word_step_kernel(
    int t, int par, const _Float16* __restrict__ Xf,
    const _Float16* __restrict__ Xb, const _Float16* __restrict__ WhhPf,
    const _Float16* __restrict__ WhhPb, _Float16* __restrict__ hbuf,
    float* __restrict__ cbuf, const int* __restrict__ words_num) {
  int lane = threadIdx.x & 31;
  int wid = (blockIdx.x * blockDim.x + threadIdx.x) >> 5;  // 0..255
  int dir = wid >> 7;
  int rem = wid & 127;
  int mt = rem >> 5;   // 0..3
  int ht = rem & 31;   // 0..31
  const _Float16* Xt = (dir ? Xb : Xf) + (size_t)t * Bn * G4;
  const _Float16* Wp = dir ? WhhPb : WhhPf;
  const _Float16* hin = hbuf + (size_t)(par * 2 + dir) * Bn * Hn;
  _Float16* hout = hbuf + (size_t)((1 - par) * 2 + dir) * Bn * Hn;
  float* cst = cbuf + (size_t)dir * Bn * Hn;

  v8f a0 = {}, a1 = {}, a2 = {}, a3 = {};
  for (int kt = 0; kt < Hn / 32; ++kt) {
    v16h a = load_a(hin, Hn, mt * 16, kt * 32, lane);
    a0 = wmma_f16(a, load_b(Wp, G4 / 16, kt, 0 * 32 + ht, lane), a0);
    a1 = wmma_f16(a, load_b(Wp, G4 / 16, kt, 1 * 32 + ht, lane), a1);
    a2 = wmma_f16(a, load_b(Wp, G4 / 16, kt, 2 * 32 + ht, lane), a2);
    a3 = wmma_f16(a, load_b(Wp, G4 / 16, kt, 3 * 32 + ht, lane), a3);
  }
  int nloc = ht * 16 + (lane & 15);
#pragma unroll
  for (int r = 0; r < 8; ++r) {
    int m = mt * 16 + r + 8 * (lane >> 4);
    bool live = t < words_num[m];
    float iv = sigmoid_f(a0[r] + (float)Xt[(size_t)m * G4 + nloc]);
    float fv = sigmoid_f(a1[r] + (float)Xt[(size_t)m * G4 + Hn + nloc]);
    float gv = tanh_f(a2[r] + (float)Xt[(size_t)m * G4 + 2 * Hn + nloc]);
    float ov = sigmoid_f(a3[r] + (float)Xt[(size_t)m * G4 + 3 * Hn + nloc]);
    size_t ci = (size_t)m * Hn + nloc;
    float cold = cst[ci];
    float c2 = fv * cold + iv * gv;
    float h2 = ov * tanh_f(c2);
    float hold = (float)hin[ci];
    if (live) cst[ci] = c2;
    hout[ci] = (_Float16)(live ? h2 : hold);
  }
}

// context = concat(where(len==S, h_f, 0), h_b) -> decoder h0 (f32 + f16 copies)
__global__ void make_context_kernel(const _Float16* __restrict__ hbuf,
                                    const int* __restrict__ words_num,
                                    float* __restrict__ dh32,
                                    _Float16* __restrict__ dh16) {
  int i = blockIdx.x * blockDim.x + threadIdx.x;
  if (i >= Bn * Gn) return;
  int b = i >> 10, j = i & (Gn - 1);
  float v;
  if (j < Hn) {
    v = (words_num[b] == Sn) ? (float)hbuf[(size_t)0 * Bn * Hn + (size_t)b * Hn + j]
                             : 0.f;
  } else {
    v = (float)hbuf[(size_t)1 * Bn * Hn + (size_t)b * Hn + (j - Hn)];
  }
  dh32[i] = v;
  dh16[i] = (_Float16)v;
}

// U[(t*B+b), :] = concat(Et[tok_in[b,t]], chars_enc[b,t,:])  (teacher forcing)
__global__ void build_u_kernel(const int* __restrict__ targets,
                               const float* __restrict__ Et,
                               const _Float16* __restrict__ cenc,
                               _Float16* __restrict__ U) {
  int i = blockIdx.x * blockDim.x + threadIdx.x;
  if (i >= Sn * Bn * DH) return;
  int j = i % DH;
  int m = i / DH;
  int b = m & (Bn - 1);
  int t = m >> 6;
  _Float16 v;
  if (j < Dn) {
    int tok = (t == 0) ? 0 : targets[b * Sn + (t - 1)];
    v = (_Float16)Et[(size_t)tok * Dn + j];
  } else {
    v = cenc[((size_t)b * Sn + t) * Hn + (j - Dn)];
  }
  U[i] = v;
}

// ---------------------------------------------------------------------------
// GRU decoder step (gi precomputed in GI). 256 waves: (mtile, htile 0..63).
// ---------------------------------------------------------------------------
__global__ __launch_bounds__(256) void gru_step_kernel(
    int t, int par, const _Float16* __restrict__ GIb,
    const _Float16* __restrict__ WhhP, const float* __restrict__ bhh,
    float* __restrict__ dh32, _Float16* __restrict__ dh16,
    _Float16* __restrict__ H2) {
  int lane = threadIdx.x & 31;
  int wid = (blockIdx.x * blockDim.x + threadIdx.x) >> 5;  // 0..255
  int mt = wid >> 6;
  int ht = wid & 63;
  const _Float16* hin16 = dh16 + (size_t)par * Bn * Gn;
  const float* hin32 = dh32 + (size_t)par * Bn * Gn;
  float* hout32 = dh32 + (size_t)(1 - par) * Bn * Gn;
  _Float16* hout16 = dh16 + (size_t)(1 - par) * Bn * Gn;
  const _Float16* GIt = GIb + (size_t)t * Bn * G3;

  v8f aR = {}, aZ = {}, aN = {};
  for (int kt = 0; kt < Gn / 32; ++kt) {
    v16h a = load_a(hin16, Gn, mt * 16, kt * 32, lane);
    aR = wmma_f16(a, load_b(WhhP, G3 / 16, kt, 0 * 64 + ht, lane), aR);
    aZ = wmma_f16(a, load_b(WhhP, G3 / 16, kt, 1 * 64 + ht, lane), aZ);
    aN = wmma_f16(a, load_b(WhhP, G3 / 16, kt, 2 * 64 + ht, lane), aN);
  }
  int nloc = ht * 16 + (lane & 15);
  float br = bhh[nloc], bz = bhh[Gn + nloc], bn = bhh[2 * Gn + nloc];
#pragma unroll
  for (int r = 0; r < 8; ++r) {
    int m = mt * 16 + r + 8 * (lane >> 4);
    float ir = (float)GIt[(size_t)m * G3 + nloc];
    float iz = (float)GIt[(size_t)m * G3 + Gn + nloc];
    float in_ = (float)GIt[(size_t)m * G3 + 2 * Gn + nloc];
    float rg = sigmoid_f(ir + aR[r] + br);
    float zg = sigmoid_f(iz + aZ[r] + bz);
    float ng = tanh_f(in_ + rg * (aN[r] + bn));
    size_t hi = (size_t)m * Gn + nloc;
    float hold = hin32[hi];
    float h2 = (1.f - zg) * ng + zg * hold;
    hout32[hi] = h2;
    hout16[hi] = (_Float16)h2;
    H2[((size_t)t * Bn + m) * Gn + nloc] = (_Float16)h2;
  }
}

// ---------------------------------------------------------------------------
// Host orchestration
// ---------------------------------------------------------------------------
extern "C" void kernel_launch(void* const* d_in, const int* in_sizes, int n_in,
                              void* d_out, int out_size, void* d_ws,
                              size_t ws_size, hipStream_t stream) {
  const int* words = (const int*)d_in[0];
  const int* words_num = (const int*)d_in[1];
  const int* chars = (const int*)d_in[2];
  const int* char_lens = (const int*)d_in[3];
  const int* targets = (const int*)d_in[4];
  const float* Ew = (const float*)d_in[5];
  const float* Ec = (const float*)d_in[6];
  const float* Et = (const float*)d_in[7];
  const float* wlf_Wih = (const float*)d_in[8];
  const float* wlf_Whh = (const float*)d_in[9];
  const float* wlf_b = (const float*)d_in[10];
  const float* wlb_Wih = (const float*)d_in[11];
  const float* wlb_Whh = (const float*)d_in[12];
  const float* wlb_b = (const float*)d_in[13];
  const float* cl_Wih = (const float*)d_in[14];
  const float* cl_Whh = (const float*)d_in[15];
  const float* cl_b = (const float*)d_in[16];
  const float* g_Wih = (const float*)d_in[17];
  const float* g_Whh = (const float*)d_in[18];
  const float* g_bih = (const float*)d_in[19];
  const float* g_bhh = (const float*)d_in[20];
  const float* Wo = (const float*)d_in[21];
  const float* bo = (const float*)d_in[22];

  char* ws = (char*)d_ws;
  size_t off = 0;
  auto alloc = [&](size_t bytes) -> char* {
    char* p = ws + off;
    off = (off + bytes + 255) & ~(size_t)255;
    return p;
  };
  _Float16* pCLI = (_Float16*)alloc((size_t)G4 * Dn * 2);
  _Float16* pCLH = (_Float16*)alloc((size_t)G4 * Hn * 2);
  _Float16* pWFI = (_Float16*)alloc((size_t)G4 * Dn * 2);
  _Float16* pWFH = (_Float16*)alloc((size_t)G4 * Hn * 2);
  _Float16* pWBI = (_Float16*)alloc((size_t)G4 * Dn * 2);
  _Float16* pWBH = (_Float16*)alloc((size_t)G4 * Hn * 2);
  _Float16* pGI = (_Float16*)alloc((size_t)G3 * DH * 2);
  _Float16* pGH = (_Float16*)alloc((size_t)G3 * Gn * 2);
  _Float16* pWO = (_Float16*)alloc((size_t)Cn * Gn * 2);
  _Float16* xeF = (_Float16*)alloc((size_t)Sn * Bn * Dn * 2);
  _Float16* xeB = (_Float16*)alloc((size_t)Sn * Bn * Dn * 2);
  _Float16* XF = (_Float16*)alloc((size_t)Sn * Bn * G4 * 2);
  _Float16* XB = (_Float16*)alloc((size_t)Sn * Bn * G4 * 2);
  _Float16* cenc = (_Float16*)alloc((size_t)Bn * Sn * Hn * 2);
  _Float16* hbuf = (_Float16*)alloc((size_t)2 * 2 * Bn * Hn * 2);  // [par][dir]
  float* cbuf = (float*)alloc((size_t)2 * Bn * Hn * 4);            // [dir]
  float* dh32 = (float*)alloc((size_t)2 * Bn * Gn * 4);            // [par]
  _Float16* dh16 = (_Float16*)alloc((size_t)2 * Bn * Gn * 2);      // [par]
  _Float16* U = (_Float16*)alloc((size_t)Sn * Bn * DH * 2);
  _Float16* GIbuf = (_Float16*)alloc((size_t)Sn * Bn * G3 * 2);
  _Float16* H2 = (_Float16*)alloc((size_t)Sn * Bn * Gn * 2);
  (void)ws_size;
  (void)in_sizes;
  (void)n_in;
  (void)out_size;

  auto cdiv = [](int a, int b) { return (a + b - 1) / b; };

  // --- pack all weights into WMMA B-fragment layout (f16) ---
  pack_b_kernel<<<cdiv(G4 * Dn, 256), 256, 0, stream>>>(cl_Wih, pCLI, G4, Dn);
  pack_b_kernel<<<cdiv(G4 * Hn, 256), 256, 0, stream>>>(cl_Whh, pCLH, G4, Hn);
  pack_b_kernel<<<cdiv(G4 * Dn, 256), 256, 0, stream>>>(wlf_Wih, pWFI, G4, Dn);
  pack_b_kernel<<<cdiv(G4 * Hn, 256), 256, 0, stream>>>(wlf_Whh, pWFH, G4, Hn);
  pack_b_kernel<<<cdiv(G4 * Dn, 256), 256, 0, stream>>>(wlb_Wih, pWBI, G4, Dn);
  pack_b_kernel<<<cdiv(G4 * Hn, 256), 256, 0, stream>>>(wlb_Whh, pWBH, G4, Hn);
  pack_b_kernel<<<cdiv(G3 * DH, 256), 256, 0, stream>>>(g_Wih, pGI, G3, DH);
  pack_b_kernel<<<cdiv(G3 * Gn, 256), 256, 0, stream>>>(g_Whh, pGH, G3, Gn);
  pack_b_kernel<<<cdiv(Cn * Gn, 256), 256, 0, stream>>>(Wo, pWO, Cn, Gn);

  // --- zero recurrent initial state (word h par0, word c) ---
  zero16_kernel<<<cdiv(2 * Bn * Hn, 256), 256, 0, stream>>>(hbuf, 2 * Bn * Hn);
  zero32_kernel<<<cdiv(2 * Bn * Hn, 256), 256, 0, stream>>>(cbuf, 2 * Bn * Hn);

  // --- embeddings + hoisted x-side GEMMs for the word bi-LSTM ---
  gather_xe_kernel<<<cdiv(Sn * Bn * Dn, 256), 256, 0, stream>>>(words, words_num,
                                                                Ew, xeF, xeB);
  {
    int MT = Sn * Bn / 16, NG = (G4 / 16) / 32;      // 512 x 4 blocks
    size_t shm = (size_t)16 * (Dn + 8) * 2;
    gemm_tile_kernel<<<MT * NG, 256, shm, stream>>>(xeF, pWFI, wlf_b, XF, nullptr,
                                                    Sn * Bn, G4, Dn, 0);
    gemm_tile_kernel<<<MT * NG, 256, shm, stream>>>(xeB, pWBI, wlb_b, XB, nullptr,
                                                    Sn * Bn, G4, Dn, 0);
  }

  // --- char LSTM (persistent, WMMA-dominant, ~413 GFLOP) ---
  char_lstm_kernel<<<Bn * Sn / 16, 256, 0, stream>>>(chars, char_lens, words_num,
                                                     Ec, pCLI, pCLH, cl_b, cenc);

  // --- word bi-LSTM recurrence: 128 dependent steps on the stream ---
  for (int t = 0; t < Sn; ++t)
    word_step_kernel<<<32, 256, 0, stream>>>(t, t & 1, XF, XB, pWFH, pWBH, hbuf,
                                             cbuf, words_num);
  make_context_kernel<<<cdiv(Bn * Gn, 256), 256, 0, stream>>>(hbuf, words_num,
                                                              dh32, dh16);

  // --- decoder: hoisted input GEMM (teacher forcing), then recurrence ---
  build_u_kernel<<<cdiv(Sn * Bn * DH, 256), 256, 0, stream>>>(targets, Et, cenc, U);
  {
    int MT = Sn * Bn / 16, NG = (G3 / 16) / 32;      // 512 x 6 blocks
    size_t shm = (size_t)16 * (DH + 8) * 2;
    gemm_tile_kernel<<<MT * NG, 256, shm, stream>>>(U, pGI, g_bih, GIbuf, nullptr,
                                                    Sn * Bn, G3, DH, 0);
  }
  for (int t = 0; t < Sn; ++t)
    gru_step_kernel<<<32, 256, 0, stream>>>(t, t & 1, GIbuf, pGH, g_bhh, dh32,
                                            dh16, H2);

  // --- deferred output projection: one big GEMM straight into d_out [B,S,C] ---
  {
    int MT = Sn * Bn / 16, NG = (Cn / 16) / 32;      // 512 x 2 blocks
    size_t shm = (size_t)16 * (Gn + 8) * 2;
    gemm_tile_kernel<<<MT * NG, 256, shm, stream>>>(H2, pWO, bo, nullptr,
                                                    (float*)d_out, Sn * Bn, Cn,
                                                    Gn, 1);
  }
}
